// Model_51677046505501
// MI455X (gfx1250) — matrix operand order repaired and checked by reading, not compiled, not run on hardware
//
#include <hip/hip_runtime.h>
#include <hip/hip_bf16.h>

// ---------------------------------------------------------------------------
// Seq2seq attention model for MI455X (gfx1250, wave32).
// Dense math on v_wmma_f32_16x16x32_bf16 (f32 accum). B-operand weights are
// pre-swizzled into fragment-native layout [kt][nt][lane][16] so each wave
// loads a full B fragment as one contiguous 32B/lane vector load. A fragments
// are vectorized (b128 loads + packed f32->bf16 converts).
// ---------------------------------------------------------------------------

#define BB 4096
#define SS 32
#define TT 32
#define VV 256
#define EE 128
#define HH 64
#define AA 64
#define G3H 192   // 3*H

typedef __attribute__((ext_vector_type(16))) __bf16 bf16x16;
typedef __attribute__((ext_vector_type(8)))  __bf16 bf16x8;
typedef __attribute__((ext_vector_type(4)))  __bf16 bf16x4;
typedef __attribute__((ext_vector_type(8)))  float  floatx8;
typedef __attribute__((ext_vector_type(4)))  float  f32x4;

__device__ __forceinline__ floatx8 wmma_bf16(const bf16x16& a, const bf16x16& b, const floatx8& c) {
    return __builtin_amdgcn_wmma_f32_16x16x32_bf16(false, a, false, b, (short)0, c, false, false);
}

// ---------------- WMMA fragment builders (ISA 7.12.2 wave32 layouts) -------
// A 16x32 bf16: lane (grp=lane>>4, lr=lane&15) holds row M=lr,
//   elems 0..7  -> K = k0 + grp*8 + i ; elems 8..15 -> K = k0 + 16 + grp*8 + i
__device__ __forceinline__ void frag_a_f32(bf16x16& a, const float* base, int ld, int lane) {
    int grp = lane >> 4, lr = lane & 15;
    const float* p = base + lr * ld + grp * 8;          // 16B-aligned by construction
    f32x4 u0 = *(const f32x4*)(p);
    f32x4 u1 = *(const f32x4*)(p + 4);
    f32x4 u2 = *(const f32x4*)(p + 16);
    f32x4 u3 = *(const f32x4*)(p + 20);
#pragma unroll
    for (int i = 0; i < 4; ++i) {
        a[i]      = (__bf16)u0[i];
        a[4 + i]  = (__bf16)u1[i];
        a[8 + i]  = (__bf16)u2[i];
        a[12 + i] = (__bf16)u3[i];
    }
}
__device__ __forceinline__ void frag_a_bf16(bf16x16& a, const __bf16* base, int ld, int lane) {
    int grp = lane >> 4, lr = lane & 15;
    const __bf16* p = base + (size_t)lr * ld + grp * 8; // 16B-aligned (ld multiple of 32)
    bf16x8 lo = *(const bf16x8*)(p);
    bf16x8 hi = *(const bf16x8*)(p + 16);
    a = __builtin_shufflevector(lo, hi, 0, 1, 2, 3, 4, 5, 6, 7, 8, 9, 10, 11, 12, 13, 14, 15);
}
// B fragment from swizzled weights: tile base -> 32 contiguous bytes per lane
__device__ __forceinline__ bf16x16 frag_b_swz(const __bf16* tileBase, int lane) {
    return *(const bf16x16*)(tileBase + (size_t)lane * 16);
}

__device__ __forceinline__ float sigm(float x) { return 1.f / (1.f + __expf(-x)); }

// ---------------- setup kernels --------------------------------------------
// Swizzle W[K,N] (f32) -> bf16 fragment layout out[kt][nt][lane][j]
__global__ void k_swzW(const float* __restrict__ W, __bf16* __restrict__ out, long total, int N) {
    long i = (long)blockIdx.x * blockDim.x + threadIdx.x;
    if (i >= total) return;
    int j    = (int)(i & 15);
    int lane = (int)((i >> 4) & 31);
    long tile = i >> 9;
    int nnt = N >> 4;
    int nt = (int)(tile % nnt), kt = (int)(tile / nnt);
    out[i] = (__bf16)W[(size_t)(kt * 32 + (j >> 3) * 16 + (lane >> 4) * 8 + (j & 7)) * N + nt * 16 + (lane & 15)];
}

__global__ void k_cvt_bf16v4(const float* __restrict__ in, __bf16* __restrict__ out, long n4) {
    long i = (long)blockIdx.x * blockDim.x + threadIdx.x;
    if (i >= n4) return;
    f32x4 v = *(const f32x4*)&in[i * 4];
    *(bf16x4*)&out[i * 4] = __builtin_convertvector(v, bf16x4);
}

__global__ void k_embed(const int* __restrict__ ids, const float* __restrict__ tab,
                        __bf16* __restrict__ out, long total4) {
    long i = (long)blockIdx.x * blockDim.x + threadIdx.x;
    if (i >= total4) return;
    long base = i * 4;
    long w = base / EE; int e = (int)(base % EE);
    f32x4 v = *(const f32x4*)&tab[(size_t)ids[w] * EE + e];
    *(bf16x4*)&out[base] = __builtin_convertvector(v, bf16x4);
}

// ---------------- generic WMMA GEMM: C[M,N] = A[M,K]@W[K,N] + bias ---------
// grid = M/64 blocks, 128 threads; each wave owns a 16-row m-tile, preloads
// all A fragments, sweeps every n-tile (A-reuse in registers, B swizzled).
template <int K, int N>
__global__ __launch_bounds__(128) void k_gemm_bf16(
    const __bf16* __restrict__ A, const __bf16* __restrict__ Wsw,
    const float* __restrict__ bias, float* __restrict__ C) {
    constexpr int NKT = K / 32, NNT = N / 16;
    int lane = threadIdx.x & 31, wave = threadIdx.x >> 5;
    int grp = lane >> 4, lr = lane & 15;
    int m0 = (blockIdx.x * 4 + wave) * 16;
    bf16x16 aF[NKT];
#pragma unroll
    for (int kt = 0; kt < NKT; ++kt)
        frag_a_bf16(aF[kt], A + (size_t)m0 * K + kt * 32, K, lane);
#pragma unroll
    for (int nt = 0; nt < NNT; ++nt) {
        floatx8 c = {};
#pragma unroll
        for (int kt = 0; kt < NKT; ++kt) {
            bf16x16 b = frag_b_swz(Wsw + ((size_t)kt * NNT + nt) * 512, lane);
            c = wmma_bf16(aF[kt], b, c);
        }
        float bv = bias[nt * 16 + lr];
#pragma unroll
        for (int v = 0; v < 8; ++v)
            C[(size_t)(m0 + v + grp * 8) * N + nt * 16 + lr] = c[v] + bv;
    }
}

// ---------------- persistent GRU encoder (one direction) -------------------
#define HLD 72
#define PLD 200
__global__ __launch_bounds__(128) void k_gru(
    const float* __restrict__ xp, const __bf16* __restrict__ Usw,
    const float* __restrict__ bU, float* __restrict__ encoded, int backward) {
    __shared__ float hs[16 * HLD];
    __shared__ float hps[16 * PLD];
    int tid = threadIdx.x, lane = tid & 31, wave = tid >> 5;
    int grp = lane >> 4, lr = lane & 15;
    int b0 = blockIdx.x * 16;
    for (int i = tid; i < 16 * HLD; i += 128) hs[i] = 0.f;
    __syncthreads();
    for (int s = 0; s < SS; ++s) {
        int sIdx = backward ? (SS - 1 - s) : s;
        // hp = h @ U : 12 n-tiles (3/wave), K=64 (2 chunks). A frags shared per wave.
        bf16x16 a0, a1;
        frag_a_f32(a0, &hs[0], HLD, lane);
        frag_a_f32(a1, &hs[32], HLD, lane);
#pragma unroll
        for (int j = 0; j < 3; ++j) {
            int nt = wave * 3 + j;
            floatx8 c = {};
            c = wmma_bf16(a0, frag_b_swz(Usw + (size_t)(0 * 12 + nt) * 512, lane), c);
            c = wmma_bf16(a1, frag_b_swz(Usw + (size_t)(1 * 12 + nt) * 512, lane), c);
#pragma unroll
            for (int v = 0; v < 8; ++v) hps[(v + grp * 8) * PLD + nt * 16 + lr] = c[v];
        }
        __syncthreads();
        // gates
        for (int e = tid; e < 16 * HH; e += 128) {
            int r = e >> 6, hh = e & 63;
            const float* xr = &xp[((size_t)(b0 + r) * SS + sIdx) * G3H];
            float xz = xr[hh], xrr = xr[64 + hh], xh = xr[128 + hh];
            float hz = hps[r * PLD + hh]        + bU[hh];
            float hr = hps[r * PLD + 64 + hh]   + bU[64 + hh];
            float hn = hps[r * PLD + 128 + hh]  + bU[128 + hh];
            float z  = sigm(xz + hz);
            float rr = sigm(xrr + hr);
            float n  = tanhf(xh + rr * hn);
            float hN = z * hs[r * HLD + hh] + (1.f - z) * n;
            hs[r * HLD + hh] = hN;
            size_t oi = ((size_t)(b0 + r) * SS + sIdx) * HH + hh;
            if (backward) encoded[oi] += hN; else encoded[oi] = hN;
            if (s + 1 < SS) {  // -> global_prefetch_b8
                int nIdx = backward ? (sIdx - 1) : (sIdx + 1);
                __builtin_prefetch(&xp[((size_t)(b0 + r) * SS + nIdx) * G3H + hh], 0, 1);
            }
        }
        __syncthreads();
    }
}

// ---------------- persistent attention decoder -----------------------------
#define ALD 136
__global__ __launch_bounds__(128) void k_decoder(
    const float* __restrict__ encoded, const float* __restrict__ enc_proj,
    const __bf16* __restrict__ WdSw, const float* __restrict__ attn_bd,
    const float* __restrict__ attn_v, const float* __restrict__ attn_bv,
    const __bf16* __restrict__ decWSw, const __bf16* __restrict__ decUSw,
    const float* __restrict__ dec_b, const float* __restrict__ out_W,
    const int* __restrict__ tgt, __bf16* __restrict__ hidden_bf) {
    __shared__ float hs[16 * HLD];
    __shared__ float dps[16 * HLD];
    __shared__ float wS[16 * SS];
    __shared__ float ai[16 * ALD];
    __shared__ float xps[16 * PLD];
    __shared__ float hps[16 * PLD];
    __shared__ float vS[AA], bdS[AA], dbS[2 * G3H];
    __shared__ float bvS;
    int tid = threadIdx.x, lane = tid & 31, wave = tid >> 5;
    int grp = lane >> 4, lr = lane & 15;
    int b0 = blockIdx.x * 16;
    if (tid < AA) { vS[tid] = attn_v[tid]; bdS[tid] = attn_bd[tid]; }
    for (int i = tid; i < 2 * G3H; i += 128) dbS[i] = dec_b[i];
    if (tid == 0) bvS = attn_bv[0];
    for (int e = tid; e < 16 * HH; e += 128) {            // h0 = encoded[:,0,:]
        int r = e >> 6, hh = e & 63;
        hs[r * HLD + hh] = encoded[(size_t)(b0 + r) * SS * HH + hh];
    }
    __syncthreads();
    for (int t = 0; t < TT; ++t) {
        // dproj = h @ attn_Wd + bd   (4 n-tiles, one per wave, K=64)
        {
            bf16x16 a0, a1;
            frag_a_f32(a0, &hs[0], HLD, lane);
            frag_a_f32(a1, &hs[32], HLD, lane);
            floatx8 c = {};
            c = wmma_bf16(a0, frag_b_swz(WdSw + (size_t)(0 * 4 + wave) * 512, lane), c);
            c = wmma_bf16(a1, frag_b_swz(WdSw + (size_t)(1 * 4 + wave) * 512, lane), c);
#pragma unroll
            for (int v = 0; v < 8; ++v)
                dps[(v + grp * 8) * HLD + wave * 16 + lr] = c[v] + bdS[wave * 16 + lr];
        }
        __syncthreads();
        // scores: logit[r][s] = sum_a tanh(dproj[r][a] + enc_proj[r,s,a]) * v[a] + bv
        for (int e = tid; e < 16 * SS; e += 128) {
            int r = e >> 5, s = e & 31;
            const f32x4* ep = (const f32x4*)&enc_proj[((size_t)(b0 + r) * SS + s) * AA];
            const f32x4* dp = (const f32x4*)&dps[r * HLD];
            float acc = bvS;
#pragma unroll 4
            for (int q = 0; q < AA / 4; ++q) {
                f32x4 ev = ep[q], dv = dp[q], vv = *(const f32x4*)&vS[q * 4];
                acc += tanhf(dv[0] + ev[0]) * vv[0] + tanhf(dv[1] + ev[1]) * vv[1] +
                       tanhf(dv[2] + ev[2]) * vv[2] + tanhf(dv[3] + ev[3]) * vv[3];
            }
            wS[r * SS + s] = acc;
        }
        __syncthreads();
        // softmax over S per row
        if (tid < 16) {
            float mx = -1e30f;
            for (int s = 0; s < SS; ++s) mx = fmaxf(mx, wS[tid * SS + s]);
            float sum = 0.f;
            for (int s = 0; s < SS; ++s) { float ex = __expf(wS[tid * SS + s] - mx); wS[tid * SS + s] = ex; sum += ex; }
            float inv = 1.f / sum;
            for (int s = 0; s < SS; ++s) wS[tid * SS + s] *= inv;
        }
        __syncthreads();
        // ctx + tied embedding -> att_in = [x | ctx]
        for (int e = tid; e < 16 * HH; e += 128) {
            int r = e >> 6, hh = e & 63;
            const float* enc = &encoded[(size_t)(b0 + r) * SS * HH + hh];
            float acc = 0.f;
            for (int s = 0; s < SS; ++s) acc += wS[r * SS + s] * enc[(size_t)s * HH];
            ai[r * ALD + HH + hh] = acc;
            int tok = (t == 0) ? 1 : tgt[(size_t)(b0 + r) * TT + (t - 1)];
            ai[r * ALD + hh] = out_W[(size_t)hh * VV + tok] * 8.0f;   // sqrt(H)=8
        }
        __syncthreads();
        // xp = att_in @ dec_W (K=128) ; hp = h @ dec_U (K=64) ; 3 n-tiles/wave
        {
            bf16x16 aw[4], ah[2];
#pragma unroll
            for (int kt = 0; kt < 4; ++kt) frag_a_f32(aw[kt], &ai[kt * 32], ALD, lane);
            frag_a_f32(ah[0], &hs[0], HLD, lane);
            frag_a_f32(ah[1], &hs[32], HLD, lane);
#pragma unroll
            for (int j = 0; j < 3; ++j) {
                int nt = wave * 3 + j;
                floatx8 c = {};
#pragma unroll
                for (int kt = 0; kt < 4; ++kt)
                    c = wmma_bf16(aw[kt], frag_b_swz(decWSw + ((size_t)kt * 12 + nt) * 512, lane), c);
#pragma unroll
                for (int v = 0; v < 8; ++v) xps[(v + grp * 8) * PLD + nt * 16 + lr] = c[v];
                floatx8 c2 = {};
#pragma unroll
                for (int kt = 0; kt < 2; ++kt)
                    c2 = wmma_bf16(ah[kt], frag_b_swz(decUSw + ((size_t)kt * 12 + nt) * 512, lane), c2);
#pragma unroll
                for (int v = 0; v < 8; ++v) hps[(v + grp * 8) * PLD + nt * 16 + lr] = c2[v];
            }
        }
        __syncthreads();
        // GRU gates
        for (int e = tid; e < 16 * HH; e += 128) {
            int r = e >> 6, hh = e & 63;
            float xz = xps[r * PLD + hh]       + dbS[hh];
            float xr = xps[r * PLD + 64 + hh]  + dbS[64 + hh];
            float xh = xps[r * PLD + 128 + hh] + dbS[128 + hh];
            float hz = hps[r * PLD + hh]       + dbS[G3H + hh];
            float hr = hps[r * PLD + 64 + hh]  + dbS[G3H + 64 + hh];
            float hn = hps[r * PLD + 128 + hh] + dbS[G3H + 128 + hh];
            float z  = sigm(xz + hz);
            float rr = sigm(xr + hr);
            float n  = tanhf(xh + rr * hn);
            float hN = z * hs[r * HLD + hh] + (1.f - z) * n;
            hs[r * HLD + hh] = hN;
            hidden_bf[((size_t)(b0 + r) * TT + t) * HH + hh] = (__bf16)hN;
        }
        __syncthreads();
    }
}

// ---------------------------------------------------------------------------
extern "C" void kernel_launch(void* const* d_in, const int* in_sizes, int n_in,
                              void* d_out, int out_size, void* d_ws, size_t ws_size,
                              hipStream_t stream) {
    const int*   src_ids = (const int*)  d_in[0];
    const int*   tgt_ids = (const int*)  d_in[1];
    const float* src_emb = (const float*)d_in[2];
    const float* enc_Wf  = (const float*)d_in[3];
    const float* enc_Uf  = (const float*)d_in[4];
    const float* enc_bf  = (const float*)d_in[5];
    const float* enc_Wb  = (const float*)d_in[6];
    const float* enc_Ub  = (const float*)d_in[7];
    const float* enc_bb  = (const float*)d_in[8];
    const float* attn_We = (const float*)d_in[9];
    const float* attn_be = (const float*)d_in[10];
    const float* attn_Wd = (const float*)d_in[11];
    const float* attn_bd = (const float*)d_in[12];
    const float* attn_v  = (const float*)d_in[13];
    const float* attn_bv = (const float*)d_in[14];
    const float* dec_W   = (const float*)d_in[15];
    const float* dec_U   = (const float*)d_in[16];
    const float* dec_b   = (const float*)d_in[17];
    const float* out_W   = (const float*)d_in[18];
    const float* out_b   = (const float*)d_in[19];
    float* logits = (float*)d_out;

    const long BS = (long)BB * SS;          // 131072
    char* ws = (char*)d_ws; size_t off = 0;
    auto alloc = [&](size_t bytes) { void* p = ws + off; off = (off + bytes + 255) & ~(size_t)255; return p; };

    __bf16* embBf   = (__bf16*)alloc(BS * EE * sizeof(__bf16));
    float*  xp      = (float*) alloc(BS * G3H * sizeof(float));   // reused fwd/bwd
    float*  enc     = (float*) alloc(BS * HH * sizeof(float));
    __bf16* encBf   = (__bf16*)alloc(BS * HH * sizeof(__bf16));
    float*  encProj = (float*) alloc(BS * AA * sizeof(float));
    __bf16* hidBf   = (__bf16*)alloc((long)BB * TT * HH * sizeof(__bf16));
    __bf16* WfS = (__bf16*)alloc(EE * G3H * 2);      // all swizzled bf16 weights
    __bf16* UfS = (__bf16*)alloc(HH * G3H * 2);
    __bf16* WbS = (__bf16*)alloc(EE * G3H * 2);
    __bf16* UbS = (__bf16*)alloc(HH * G3H * 2);
    __bf16* WeS = (__bf16*)alloc(HH * AA * 2);
    __bf16* WdS = (__bf16*)alloc(HH * AA * 2);
    __bf16* dWS = (__bf16*)alloc(2 * HH * G3H * 2);
    __bf16* dUS = (__bf16*)alloc(HH * G3H * 2);
    __bf16* oWS = (__bf16*)alloc(HH * VV * 2);

    auto swz = [&](const float* src, __bf16* dst, long K, long N) {
        long n = K * N;
        k_swzW<<<dim3((unsigned)((n + 255) / 256)), dim3(256), 0, stream>>>(src, dst, n, (int)N);
    };
    swz(enc_Wf, WfS, EE, G3H);  swz(enc_Uf, UfS, HH, G3H);
    swz(enc_Wb, WbS, EE, G3H);  swz(enc_Ub, UbS, HH, G3H);
    swz(attn_We, WeS, HH, AA);  swz(attn_Wd, WdS, HH, AA);
    swz(dec_W, dWS, 2 * HH, G3H); swz(dec_U, dUS, HH, G3H);
    swz(out_W, oWS, HH, VV);

    { // embedding gather -> bf16 (4 elems/thread)
        long n4 = BS * EE / 4;
        k_embed<<<dim3((unsigned)((n4 + 255) / 256)), dim3(256), 0, stream>>>(src_ids, src_emb, embBf, n4);
    }
    // forward direction
    k_gemm_bf16<EE, G3H><<<dim3(BS / 64), dim3(128), 0, stream>>>(embBf, WfS, enc_bf, xp);
    k_gru<<<dim3(BB / 16), dim3(128), 0, stream>>>(xp, UfS, enc_bf + G3H, enc, 0);
    // backward direction (reuses xp buffer; stream-ordered)
    k_gemm_bf16<EE, G3H><<<dim3(BS / 64), dim3(128), 0, stream>>>(embBf, WbS, enc_bb, xp);
    k_gru<<<dim3(BB / 16), dim3(128), 0, stream>>>(xp, UbS, enc_bb + G3H, enc, 1);
    // enc_proj = encoded @ attn_We + attn_be
    {
        long n4 = BS * HH / 4;
        k_cvt_bf16v4<<<dim3((unsigned)((n4 + 255) / 256)), dim3(256), 0, stream>>>(enc, encBf, n4);
    }
    k_gemm_bf16<HH, AA><<<dim3(BS / 64), dim3(128), 0, stream>>>(encBf, WeS, attn_be, encProj);
    // decoder
    k_decoder<<<dim3(BB / 16), dim3(128), 0, stream>>>(enc, encProj, WdS, attn_bd, attn_v, attn_bv,
                                                       dWS, dUS, dec_b, out_W, tgt_ids, hidBf);
    // logits = hidden @ out_W + out_b
    k_gemm_bf16<HH, VV><<<dim3((long)BB * TT / 64), dim3(128), 0, stream>>>(hidBf, oWS, out_b, logits);
    (void)in_sizes; (void)n_in; (void)out_size; (void)ws_size;
}
